// GatedTitansMAC_1477468750475
// MI455X (gfx1250) — matrix operand order repaired and checked by reading, not compiled, not run on hardware
//
#include <hip/hip_runtime.h>
#include <hip/hip_bf16.h>
#include <math.h>

// ---------------------------------------------------------------------------
// Types for CDNA5 WMMA
// ---------------------------------------------------------------------------
typedef __attribute__((ext_vector_type(16))) __bf16 v16bf;
typedef __attribute__((ext_vector_type(8)))  float  v8f;
typedef __attribute__((ext_vector_type(2)))  __bf16 v2bf;
typedef __attribute__((ext_vector_type(2)))  float  v2f;

// Native f32->bf16 (RNE). Scalar form: one v_cvt_pk_bf16_f32.
__device__ __forceinline__ unsigned short f2bf(float f) {
    union { __bf16 h; unsigned short u; } c;
    c.h = (__bf16)f;
    return c.u;
}
// Packed pair: __builtin_convertvector lowers to a single
// v_cvt_pk_bf16_f32 dst, a, b (no shift/or assembly).
__device__ __forceinline__ unsigned int pack2bf(float a, float b) {
    v2f f{a, b};
    v2bf h = __builtin_convertvector(f, v2bf);
    union { v2bf v; unsigned int u; } c;
    c.v = h;
    return c.u;
}
__device__ __forceinline__ float sigmoidf_(float x) { return 1.0f / (1.0f + __expf(-x)); }

// ---------------------------------------------------------------------------
// Generic bf16-WMMA GEMM (all variants compile-time specialized):
//   C[M,N] = op(A)[M,K] * op(B)[K,N] (+ bias) (+= C)
//   TRANSA: A stored [K,M] row-major.  TRANSB: B stored [N,K] row-major.
//   ROWSCALE: scale A row m by rowScale[m].  ACC: C += result.
// Tile 128x128x32, 256 threads (8 waves), wave -> 64x32 (4x2 WMMA tiles).
// A in LDS: [m][k] ushorts (padded stride 40) -> frag = 2x ds_load_b128.
// B in LDS: K-pair-packed dwords Bs2[k/2][n] (padded stride 132) ->
//           frag dword i = Bs2[i + 4*(lane>=16) + 4*(i>=4)][col] (ds_load_2addr_b32 pairs).
// Software pipeline: next K-chunk loaded to registers while WMMAs run.
// Split-K via blockIdx.z: slice covers kLen of K, partial at C + z*M*N.
// Assumes Kdim % 32 == 0 and all dims % 4 == 0 (true here).
// ---------------------------------------------------------------------------
#define BMT 128
#define BNT 128
#define BKT 32
#define LDA_S 40     // padded ushort stride for As rows
#define LDB2_S 132   // padded uint stride for Bs2 rows (16 rows of k-pairs)
#define KSPLIT 32

template<int TRANSA, int TRANSB, bool ROWSCALE, bool BIAS, bool ACC>
__launch_bounds__(256)
__global__ void wmma_gemm(const float* __restrict__ A,
                          const float* __restrict__ Bm,
                          float* __restrict__ C,
                          const float* __restrict__ bias,
                          const float* __restrict__ rowScale,
                          int Mdim, int Ndim, int Kdim, int kLen)
{
    __shared__ __align__(16) unsigned short As[BMT * LDA_S];
    __shared__ __align__(16) unsigned int   Bs2[16 * LDB2_S];
    unsigned short* Bs2u = (unsigned short*)Bs2;

    const int tid   = threadIdx.x;
    const int lane  = tid & 31;
    const int wave  = tid >> 5;
    const int waveM = (wave >> 2) * 64;   // 0 or 64
    const int waveN = (wave & 3) * 32;    // 0,32,64,96
    const int blockM = blockIdx.y * BMT;
    const int blockN = blockIdx.x * BNT;
    const int nloc = lane & 15;
    const int half = lane >> 4;

    const int kStart = blockIdx.z * kLen;
    const int kEnd   = (kStart + kLen < Kdim) ? kStart + kLen : Kdim;
    C += (size_t)blockIdx.z * ((size_t)Mdim * Ndim);

    v8f acc[4][2];
    #pragma unroll
    for (int i = 0; i < 4; i++)
        #pragma unroll
        for (int j = 0; j < 2; j++)
            #pragma unroll
            for (int r = 0; r < 8; r++) {
                if constexpr (ACC) {
                    int gm = blockM + waveM + i * 16 + r + 8 * half;
                    int gn = blockN + waveN + j * 16 + nloc;
                    acc[i][j][r] = (gm < Mdim && gn < Ndim) ? C[(size_t)gm * Ndim + gn] : 0.0f;
                } else {
                    acc[i][j][r] = 0.0f;
                }
            }

    float4 aReg[4], bReg[4];

    auto loadA = [&](int k0) {
        #pragma unroll
        for (int i = 0; i < 4; i++) {
            float4 v{0.f, 0.f, 0.f, 0.f};
            if constexpr (TRANSA == 0) {
                int e = (tid + i * 256) * 4;
                int r = e >> 5, c = e & 31;          // 4 consecutive K
                int gm = blockM + r;
                if (gm < Mdim) {
                    v = *(const float4*)&A[(size_t)gm * Kdim + (k0 + c)];
                    if constexpr (ROWSCALE) {
                        float s = rowScale[gm];
                        v.x *= s; v.y *= s; v.z *= s; v.w *= s;
                    }
                }
            } else {
                int e = (tid + i * 256) * 4;
                int kk = e >> 7, m = e & 127;        // 4 consecutive M (contiguous mem)
                int gm = blockM + m;
                if (gm + 3 < Mdim) {
                    v = *(const float4*)&A[(size_t)(k0 + kk) * Mdim + gm];
                } else if (gm < Mdim) {
                    float* pv = &v.x;
                    for (int q = 0; q < 4; q++)
                        if (gm + q < Mdim) pv[q] = A[(size_t)(k0 + kk) * Mdim + gm + q];
                }
            }
            aReg[i] = v;
        }
    };
    auto loadB = [&](int k0) {
        #pragma unroll
        for (int i = 0; i < 4; i++) {
            float4 v{0.f, 0.f, 0.f, 0.f};
            if constexpr (TRANSB == 0) {
                int e = (tid + i * 256) * 4;
                int r = e >> 7, c = e & 127;         // 4 consecutive N (contiguous mem)
                int gn = blockN + c;
                if (gn < Ndim)
                    v = *(const float4*)&Bm[(size_t)(k0 + r) * Ndim + gn];
            } else {
                int e = (tid + i * 256) * 4;
                int n = e >> 5, kk = e & 31;         // 4 consecutive K (contiguous mem)
                int gn = blockN + n;
                if (gn < Ndim)
                    v = *(const float4*)&Bm[(size_t)gn * Kdim + (k0 + kk)];
            }
            bReg[i] = v;
        }
    };
    auto storeA = [&]() {
        #pragma unroll
        for (int i = 0; i < 4; i++) {
            int e = (tid + i * 256) * 4;
            if constexpr (TRANSA == 0) {
                int r = e >> 5, c = e & 31;
                uint2 p{pack2bf(aReg[i].x, aReg[i].y), pack2bf(aReg[i].z, aReg[i].w)};
                *(uint2*)&As[r * LDA_S + c] = p;     // 2x v_cvt_pk + ds_store_b64
            } else {
                int kk = e >> 7, m = e & 127;
                As[(m + 0) * LDA_S + kk] = f2bf(aReg[i].x);
                As[(m + 1) * LDA_S + kk] = f2bf(aReg[i].y);
                As[(m + 2) * LDA_S + kk] = f2bf(aReg[i].z);
                As[(m + 3) * LDA_S + kk] = f2bf(aReg[i].w);
            }
        }
    };
    auto storeB = [&]() {
        #pragma unroll
        for (int i = 0; i < 4; i++) {
            int e = (tid + i * 256) * 4;
            if constexpr (TRANSB == 0) {
                int r = e >> 7, c = e & 127;         // fixed k, 4 consecutive n
                int row = r >> 1, hw = r & 1;
                Bs2u[((row * LDB2_S) + c + 0) * 2 + hw] = f2bf(bReg[i].x);
                Bs2u[((row * LDB2_S) + c + 1) * 2 + hw] = f2bf(bReg[i].y);
                Bs2u[((row * LDB2_S) + c + 2) * 2 + hw] = f2bf(bReg[i].z);
                Bs2u[((row * LDB2_S) + c + 3) * 2 + hw] = f2bf(bReg[i].w);
            } else {
                int n = e >> 5, kk = e & 31;         // fixed n, 4 consecutive k
                const float* pv = &bReg[i].x;
                #pragma unroll
                for (int q = 0; q < 4; q++) {
                    int k = kk + q;
                    Bs2u[(((k >> 1) * LDB2_S) + n) * 2 + (k & 1)] = f2bf(pv[q]);
                }
            }
        }
    };

    loadA(kStart);
    loadB(kStart);

    for (int k0 = kStart; k0 < kEnd; k0 += BKT) {
        storeA();
        storeB();
        __syncthreads();

        // issue next chunk's global loads before consuming LDS (latency hiding)
        if (k0 + BKT < kEnd) { loadA(k0 + BKT); loadB(k0 + BKT); }

        union FragU { v16bf v; unsigned short u[16]; unsigned int d[8]; };
        FragU a[4], b[2];
        #pragma unroll
        for (int t = 0; t < 4; t++) {
            int row = waveM + t * 16 + nloc;         // A: M = lane%16
            #pragma unroll
            for (int i = 0; i < 8; i++) {
                int kk = (i < 4 ? 2 * i : 16 + 2 * (i - 4)) + 8 * half;
                a[t].d[i] = *(const unsigned int*)&As[row * LDA_S + kk]; // merges to ds_load_b128
            }
        }
        #pragma unroll
        for (int t = 0; t < 2; t++) {
            int col = waveN + t * 16 + nloc;         // B: N = lane%16
            #pragma unroll
            for (int i = 0; i < 8; i++)              // dword i: K = 2i,2i+1 (+8 for half, +8 for i>=4)
                b[t].d[i] = Bs2[(i + 4 * half + (i < 4 ? 0 : 4)) * LDB2_S + col];
        }

        #pragma unroll
        for (int i = 0; i < 4; i++)
            #pragma unroll
            for (int j = 0; j < 2; j++)
                acc[i][j] = __builtin_amdgcn_wmma_f32_16x16x32_bf16(
                    false, a[i].v, false, b[j].v, (short)0, acc[i][j], false, false);

        __syncthreads();
    }

    // epilogue: f32 C/D layout: vgpr r -> M = r + 8*(lane>=16), N = lane%16
    float bv[2] = {0.f, 0.f};
    if constexpr (BIAS) {
        #pragma unroll
        for (int j = 0; j < 2; j++) {
            int gn = blockN + waveN + j * 16 + nloc;
            bv[j] = (gn < Ndim) ? bias[gn] : 0.f;
        }
    }
    #pragma unroll
    for (int i = 0; i < 4; i++)
        #pragma unroll
        for (int j = 0; j < 2; j++)
            #pragma unroll
            for (int r = 0; r < 8; r++) {
                int gm = blockM + waveM + i * 16 + r + 8 * half;
                int gn = blockN + waveN + j * 16 + nloc;
                if (gm < Mdim && gn < Ndim)
                    C[(size_t)gm * Ndim + gn] = acc[i][j][r] + bv[j];
            }
}

// ---------------------------------------------------------------------------
// Elementwise / reduction kernels
// ---------------------------------------------------------------------------
__global__ void silu_kernel(const float* __restrict__ x, float* __restrict__ y, int n) {
    int i = blockIdx.x * blockDim.x + threadIdx.x;
    if (i < n) { float v = x[i]; y[i] = v * sigmoidf_(v); }
}

// dpre = dact * silu'(hpre);  silu'(x) = s*(1 + x*(1-s))
__global__ void silu_bwd_kernel(const float* __restrict__ dact,
                                const float* __restrict__ hpre,
                                float* __restrict__ dpre, int n) {
    int i = blockIdx.x * blockDim.x + threadIdx.x;
    if (i < n) {
        float x = hpre[i];
        float s = sigmoidf_(x);
        dpre[i] = dact[i] * (s * (1.0f + x * (1.0f - s)));
    }
}

// per-row: per_sample[b] = mean_H (pred-v)^2 ; dpred = 2*(pred-v)/(B*H)
__global__ void loss_dpred_kernel(const float* __restrict__ pred,
                                  const float* __restrict__ v,
                                  float* __restrict__ dpred,
                                  float* __restrict__ per_sample,
                                  int Bn, int Hn) {
    __shared__ float red[256];
    int b = blockIdx.x;
    float scale = 2.0f / ((float)Bn * (float)Hn);
    float s = 0.0f;
    for (int h = threadIdx.x; h < Hn; h += 256) {
        size_t idx = (size_t)b * Hn + h;
        float d = pred[idx] - v[idx];
        dpred[idx] = scale * d;
        s += d * d;
    }
    red[threadIdx.x] = s; __syncthreads();
    for (int off = 128; off > 0; off >>= 1) {
        if (threadIdx.x < off) red[threadIdx.x] += red[threadIdx.x + off];
        __syncthreads();
    }
    if (threadIdx.x == 0) per_sample[b] = red[0] / (float)Hn;
}

// gate[b] = sigmoid(hidden[b]·w_gate + b_gate + per_sample[b]) * 0.99^step
__global__ void gate_kernel(const float* __restrict__ hidden,
                            const float* __restrict__ w_gate,
                            const float* __restrict__ b_gate,
                            const float* __restrict__ per_sample,
                            const int* __restrict__ step,
                            float* __restrict__ gate, int Bn, int Hn) {
    __shared__ float red[256];
    int b = blockIdx.x;
    float s = 0.0f;
    for (int h = threadIdx.x; h < Hn; h += 256)
        s += hidden[(size_t)b * Hn + h] * w_gate[h];
    red[threadIdx.x] = s; __syncthreads();
    for (int off = 128; off > 0; off >>= 1) {
        if (threadIdx.x < off) red[threadIdx.x] += red[threadIdx.x + off];
        __syncthreads();
    }
    if (threadIdx.x == 0) {
        float decay = __powf(0.99f, (float)step[0]);
        float z = red[0] + b_gate[0] + per_sample[b];
        gate[b] = sigmoidf_(z) * decay;
    }
}

// single-block deterministic mean
__global__ void mean_reduce_kernel(const float* __restrict__ src, int n,
                                   float* __restrict__ dst) {
    __shared__ float red[256];
    float s = 0.0f;
    for (int i = threadIdx.x; i < n; i += 256) s += src[i];
    red[threadIdx.x] = s; __syncthreads();
    for (int off = 128; off > 0; off >>= 1) {
        if (threadIdx.x < off) red[threadIdx.x] += red[threadIdx.x + off];
        __syncthreads();
    }
    if (threadIdx.x == 0) dst[0] = red[0] / (float)n;
}

// sum P split-K partials (deterministic order)
__global__ void reduce_split_kernel(const float* __restrict__ part,
                                    float* __restrict__ dst, int n, int P) {
    int i = blockIdx.x * blockDim.x + threadIdx.x;
    if (i < n) {
        float s = 0.0f;
        for (int p = 0; p < P; p++) s += part[(size_t)p * n + i];
        dst[i] = s;
    }
}

// S1n = 0.9*S1 + g1; M1n = M1 - 0.1*g*S1n (both memories, same elem count H*M)
__global__ void state_update_kernel(const float* __restrict__ M1,
                                    const float* __restrict__ M2,
                                    const float* __restrict__ S1,
                                    const float* __restrict__ S2,
                                    const float* __restrict__ g1,
                                    const float* __restrict__ g2,
                                    const float* __restrict__ gmean,
                                    float* __restrict__ M1n, float* __restrict__ M2n,
                                    float* __restrict__ S1n, float* __restrict__ S2n,
                                    int n) {
    int i = blockIdx.x * blockDim.x + threadIdx.x;
    if (i >= n) return;
    float g = gmean[0];
    float s1 = 0.9f * S1[i] + g1[i];
    S1n[i] = s1;
    M1n[i] = M1[i] - 0.1f * g * s1;
    float s2 = 0.9f * S2[i] + g2[i];
    S2n[i] = s2;
    M2n[i] = M2[i] - 0.1f * g * s2;
}

// ---------------------------------------------------------------------------
// Launch
// ---------------------------------------------------------------------------
extern "C" void kernel_launch(void* const* d_in, const int* in_sizes, int n_in,
                              void* d_out, int out_size, void* d_ws, size_t ws_size,
                              hipStream_t stream) {
    constexpr int Bn = 16384, Hn = 1024, Mn = 64;

    const float* hidden = (const float*)d_in[0];
    const float* W_K    = (const float*)d_in[1];
    const float* b_K    = (const float*)d_in[2];
    const float* W_V    = (const float*)d_in[3];
    const float* b_V    = (const float*)d_in[4];
    const float* M1     = (const float*)d_in[5];
    const float* M2     = (const float*)d_in[6];
    const float* S1     = (const float*)d_in[7];
    const float* S2     = (const float*)d_in[8];
    const float* w_gate = (const float*)d_in[9];
    const float* b_gate = (const float*)d_in[10];
    const float* W_out  = (const float*)d_in[11];
    const float* b_out  = (const float*)d_in[12];
    const int*   step   = (const int*)d_in[13];

    // output layout: out(B*H), loss(1), M1n(H*M), M2n(M*H), S1n(H*M), S2n(M*H)
    float* out  = (float*)d_out;
    float* loss = out + (size_t)Bn * Hn;
    float* M1n  = loss + 1;
    float* M2n  = M1n + (size_t)Hn * Mn;
    float* S1n  = M2n + (size_t)Mn * Hn;
    float* S2n  = S1n + (size_t)Hn * Mn;

    // workspace carve-up
    float* ws    = (float*)d_ws;
    float* kbuf  = ws;                             // B*H
    float* vbuf  = kbuf  + (size_t)Bn * Hn;        // B*H  (dead after loss_dpred)
    float* hpre  = vbuf  + (size_t)Bn * Hn;        // B*M
    float* act   = hpre  + (size_t)Bn * Mn;        // B*M
    float* memc  = act   + (size_t)Bn * Mn;        // B*H
    float* dpred = memc  + (size_t)Bn * Hn;        // B*H
    float* dact  = dpred + (size_t)Bn * Hn;        // B*M
    float* dpre  = dact  + (size_t)Bn * Mn;        // B*M
    float* ps    = dpre  + (size_t)Bn * Mn;        // B
    float* gate  = ps    + Bn;                     // B
    float* gmean = gate  + Bn;                     // 1
    float* g1    = gmean + 1;                      // H*M
    float* g2    = g1    + (size_t)Hn * Mn;        // M*H
    // split-K partials reuse vbuf's region (vbuf is consumed before these write)
    float* g2p   = vbuf;                           // KSPLIT * M*H
    float* g1p   = vbuf + (size_t)KSPLIT * Mn * Hn;// KSPLIT * H*M

    auto grid2 = [](int M_, int N_) { return dim3((N_ + BNT - 1) / BNT, (M_ + BMT - 1) / BMT); };
    dim3 blk(256);

    // 1) k = hidden@W_K + b_K ; v = hidden@W_V + b_V
    wmma_gemm<0,0,false,true,false><<<grid2(Bn, Hn), blk, 0, stream>>>(
        hidden, W_K, kbuf, b_K, nullptr, Bn, Hn, Hn, Hn);
    wmma_gemm<0,0,false,true,false><<<grid2(Bn, Hn), blk, 0, stream>>>(
        hidden, W_V, vbuf, b_V, nullptr, Bn, Hn, Hn, Hn);

    // 2) memory forward: hpre = k@M1 ; act = silu(hpre) ; mem_ctx = act@M2
    wmma_gemm<0,0,false,false,false><<<grid2(Bn, Mn), blk, 0, stream>>>(
        kbuf, M1, hpre, nullptr, nullptr, Bn, Mn, Hn, Hn);
    silu_kernel<<<(Bn * Mn + 255) / 256, blk, 0, stream>>>(hpre, act, Bn * Mn);
    wmma_gemm<0,0,false,false,false><<<grid2(Bn, Hn), blk, 0, stream>>>(
        act, M2, memc, nullptr, nullptr, Bn, Hn, Mn, Mn);

    // 3) loss / dpred / per-sample  (vbuf last read here)
    loss_dpred_kernel<<<Bn, blk, 0, stream>>>(memc, vbuf, dpred, ps, Bn, Hn);
    mean_reduce_kernel<<<1, blk, 0, stream>>>(ps, Bn, loss);

    // 4) grads (split-K x32 on the two K=16384 reductions)
    {
        dim3 g = grid2(Mn, Hn); g.z = KSPLIT;      // g2 = act^T @ dpred : 64x1024
        wmma_gemm<1,0,false,false,false><<<g, blk, 0, stream>>>(
            act, dpred, g2p, nullptr, nullptr, Mn, Hn, Bn, Bn / KSPLIT);
        reduce_split_kernel<<<(Mn * Hn + 255) / 256, blk, 0, stream>>>(g2p, g2, Mn * Hn, KSPLIT);
    }
    wmma_gemm<0,1,false,false,false><<<grid2(Bn, Mn), blk, 0, stream>>>(
        dpred, M2, dact, nullptr, nullptr, Bn, Mn, Hn, Hn);   // dact = dpred @ M2^T
    silu_bwd_kernel<<<(Bn * Mn + 255) / 256, blk, 0, stream>>>(dact, hpre, dpre, Bn * Mn);
    {
        dim3 g = grid2(Hn, Mn); g.z = KSPLIT;      // g1 = k^T @ dpre : 1024x64
        wmma_gemm<1,0,false,false,false><<<g, blk, 0, stream>>>(
            kbuf, dpre, g1p, nullptr, nullptr, Hn, Mn, Bn, Bn / KSPLIT);
        reduce_split_kernel<<<(Hn * Mn + 255) / 256, blk, 0, stream>>>(g1p, g1, Hn * Mn, KSPLIT);
    }

    // 5) gate + mean(gate)
    gate_kernel<<<Bn, blk, 0, stream>>>(hidden, w_gate, b_gate, ps, step, gate, Bn, Hn);
    mean_reduce_kernel<<<1, blk, 0, stream>>>(gate, Bn, gmean);

    // 6) momentum state update
    state_update_kernel<<<(Hn * Mn + 255) / 256, blk, 0, stream>>>(
        M1, M2, S1, S2, g1, g2, gmean, M1n, M2n, S1n, S2n, Hn * Mn);

    // 7) out = hidden@W_out[:H] + b_out ; out += (gate.*mem_ctx)@W_out[H:]
    wmma_gemm<0,0,false,true,false><<<grid2(Bn, Hn), blk, 0, stream>>>(
        hidden, W_out, out, b_out, nullptr, Bn, Hn, Hn, Hn);
    wmma_gemm<0,0,true,false,true><<<grid2(Bn, Hn), blk, 0, stream>>>(
        memc, W_out + (size_t)Hn * Hn, out, nullptr, gate, Bn, Hn, Hn, Hn);
}
